// Qwen3MoeAttention_88235808129237
// MI455X (gfx1250) — compile-verified, matrix operand order checked by reading, not run on hardware
//
#include <hip/hip_runtime.h>
#include <hip/hip_bf16.h>

// ---------------------------------------------------------------------------
// Qwen3-style GQA attention layer for gfx1250 (MI455X), bf16 WMMA path.
//
// Workspace layout (d_ws), total ~126 MB:
//   Xbf    [2048][2048]  bf16   hidden states
//   WqkvT  [5120][2048]  bf16   Wqkv transposed ([N][K])
//   WoT    [2048][4096]  bf16   Wo transposed ([N][K])
//   qkv    [2048][5120]  f32    QKV GEMM output
//   q_bf   [2048][32][128] bf16 normed+roped Q
//   k_bf   [4][2048][128]  bf16 normed+roped K  (== BT layout for Q@K^T)
//   vT     [4][128][2048]  bf16 V transposed    (== BT layout for P@V)
//   ctx    [2048][4096]  bf16   attention output
// ---------------------------------------------------------------------------

typedef __attribute__((ext_vector_type(16))) __bf16 v16bf;
typedef __attribute__((ext_vector_type(8)))  __bf16 v8bf;
typedef __attribute__((ext_vector_type(8)))  float  v8f;

#define S_LEN 2048
#define NHEAD 32
#define NKVH  4
#define HDIM  128
#define QKVW  5120   // (32 + 2*4) * 128
#define CTXW  4096   // 32 * 128

// round-to-nearest-even f32 -> bf16 (bit pattern as ushort)
__device__ __forceinline__ unsigned short f2bf(float f) {
  unsigned int u = __float_as_uint(f);
  u += 0x7FFFu + ((u >> 16) & 1u);
  return (unsigned short)(u >> 16);
}

// Load one 16-bit A/B WMMA fragment: per-lane two contiguous 8-element runs,
// run0 at base, run1 at base+16 elements (ISA 16-bit A/B 16x32 layout,
// caller already folded in the +8*(lane>=16) half-wave offset).
__device__ __forceinline__ v16bf load_frag(const unsigned short* base) {
  v8bf lo = *reinterpret_cast<const v8bf*>(base);
  v8bf hi = *reinterpret_cast<const v8bf*>(base + 16);
  v16bf r;
#pragma unroll
  for (int i = 0; i < 8; ++i) { r[i] = lo[i]; r[i + 8] = hi[i]; }
  return r;
}

__device__ __forceinline__ v8f wmma_bf16(v16bf a, v16bf b, v8f c) {
  return __builtin_amdgcn_wmma_f32_16x16x32_bf16(
      /*neg_a=*/false, a, /*neg_b=*/false, b,
      /*c_mod=*/(short)0, c, /*reuse_a=*/false, /*reuse_b=*/false);
}

// ---------------------------------------------------------------------------
// elementwise f32 -> bf16 convert
__global__ __launch_bounds__(256) void cvt_kernel(const float* __restrict__ src,
                                                  unsigned short* __restrict__ dst,
                                                  int n) {
  int i = blockIdx.x * blockDim.x + threadIdx.x;
  if (i < n) dst[i] = f2bf(src[i]);
}

// transpose + convert: src [rows][cols] f32 -> dst [cols][rows] bf16
__global__ __launch_bounds__(256) void cvtT_kernel(const float* __restrict__ src,
                                                   unsigned short* __restrict__ dst,
                                                   int rows, int cols) {
  int i = blockIdx.x * blockDim.x + threadIdx.x;
  if (i < rows * cols) {
    int r = i / cols, c = i - r * cols;
    dst[(size_t)c * rows + r] = f2bf(src[i]);
  }
}

// ---------------------------------------------------------------------------
// C[M][N] f32 = A[M][K] bf16  x  BT[N][K] bf16  (B pre-transposed)
// block = 256 threads = 8 waves; tile 64(M) x 128(N); wave = 16(M) x 64(N).
// K loop unrolled x2 with two alternating fragment sets: every loop-carried
// fragment is written directly by a global_load_b128, so there is no
// register-rotation (v_mov) in the steady state and loads overlap WMMAs.
// Requires K % 64 == 0 (holds for K = 2048, 4096).
__global__ __launch_bounds__(256) void gemm_bf16(const unsigned short* __restrict__ A,
                                                 const unsigned short* __restrict__ BT,
                                                 float* __restrict__ C,
                                                 int M, int N, int K) {
  const int lane  = threadIdx.x & 31;
  const int w     = threadIdx.x >> 5;
  const int wr    = w & 3;
  const int wc    = w >> 2;
  const int lhalf = lane >> 4;
  const int l16   = lane & 15;
  const int m0 = blockIdx.y * 64 + wr * 16;
  const int n0 = blockIdx.x * 128 + wc * 64;

  v8f acc[4] = {};
  const unsigned short* arow = A + (size_t)(m0 + l16) * K + 8 * lhalf;
  const unsigned short* brow[4];
#pragma unroll
  for (int t = 0; t < 4; ++t)
    brow[t] = BT + (size_t)(n0 + t * 16 + l16) * K + 8 * lhalf;

  // prologue: set0 <- fragments for k = 0
  v16bf a0 = load_frag(arow);
  v16bf b0[4];
#pragma unroll
  for (int t = 0; t < 4; ++t) b0[t] = load_frag(brow[t]);

  int k = 0;
  for (; k < K - 64; k += 64) {
    // set1 <- k+32 (overlaps with set0's WMMAs below)
    __builtin_prefetch(arow + k + 128, 0, 3);
    v16bf a1 = load_frag(arow + k + 32);
    v16bf b1[4];
#pragma unroll
    for (int t = 0; t < 4; ++t) {
      __builtin_prefetch(brow[t] + k + 128, 0, 3);
      b1[t] = load_frag(brow[t] + k + 32);
    }
#pragma unroll
    for (int t = 0; t < 4; ++t) acc[t] = wmma_bf16(a0, b0[t], acc[t]);

    // set0 <- k+64 (overlaps with set1's WMMAs below)
    a0 = load_frag(arow + k + 64);
#pragma unroll
    for (int t = 0; t < 4; ++t) b0[t] = load_frag(brow[t] + k + 64);
#pragma unroll
    for (int t = 0; t < 4; ++t) acc[t] = wmma_bf16(a1, b1[t], acc[t]);
  }

  // tail: set0 holds k = K-64; last chunk at K-32
  {
    v16bf a1 = load_frag(arow + k + 32);
    v16bf b1[4];
#pragma unroll
    for (int t = 0; t < 4; ++t) b1[t] = load_frag(brow[t] + k + 32);
#pragma unroll
    for (int t = 0; t < 4; ++t) acc[t] = wmma_bf16(a0, b0[t], acc[t]);
#pragma unroll
    for (int t = 0; t < 4; ++t) acc[t] = wmma_bf16(a1, b1[t], acc[t]);
  }

#pragma unroll
  for (int t = 0; t < 4; ++t) {
#pragma unroll
    for (int r = 0; r < 8; ++r) {
      int m = m0 + r + 8 * lhalf;        // C layout: VGPR r -> M=r (+8 for hi half)
      int n = n0 + t * 16 + l16;         //           lane%16 -> N
      C[(size_t)m * N + n] = acc[t][r];
    }
  }
}

// ---------------------------------------------------------------------------
// RMSNorm + RoPE for Q/K, transpose-convert for V.
// grid = (S, NH + 2*NKV) = (2048, 40); block = 128 (one thread per HD elem)
__global__ __launch_bounds__(128) void norm_rope_kernel(
    const float* __restrict__ qkv, const int* __restrict__ positions,
    const float* __restrict__ qw, const float* __restrict__ kw,
    unsigned short* __restrict__ q_bf,   // [S][NH][HD]
    unsigned short* __restrict__ k_bf,   // [NKV][S][HD]
    unsigned short* __restrict__ vT) {   // [NKV][HD][S]
  const int s = blockIdx.x;
  const int head = blockIdx.y;
  const int d = threadIdx.x;

  if (head >= NHEAD + NKVH) {            // V path: plain transpose convert
    int kv = head - (NHEAD + NKVH);
    float v = qkv[(size_t)s * QKVW + (NHEAD + NKVH) * HDIM + kv * HDIM + d];
    vT[((size_t)kv * HDIM + d) * S_LEN + s] = f2bf(v);
    return;
  }

  __shared__ float sh[HDIM];
  __shared__ float red[HDIM];
  const bool isq = head < NHEAD;
  const int col = isq ? head * HDIM + d : NHEAD * HDIM + (head - NHEAD) * HDIM + d;
  float x = qkv[(size_t)s * QKVW + col];

  red[d] = x * x;
  __syncthreads();
#pragma unroll
  for (int st = 64; st > 0; st >>= 1) {
    if (d < st) red[d] += red[d + st];
    __syncthreads();
  }
  float inv = rsqrtf(red[0] * (1.0f / HDIM) + 1e-6f);
  float wgt = isq ? qw[d] : kw[d];
  sh[d] = x * inv * wgt;
  __syncthreads();

  int j = d & 63;
  float p = (float)positions[s];
  // inv_freq[j] = 10000^(-2j/128); ln(10000) = 9.210340371976184
  float freq = p * __expf(-(float)(2 * j) * (9.210340371976184f / 128.0f));
  float c = __cosf(freq), sn = __sinf(freq);
  float x1 = sh[j], x2 = sh[j + 64];
  float o = (d < 64) ? (x1 * c - x2 * sn) : (x2 * c + x1 * sn);

  if (isq)
    q_bf[((size_t)s * NHEAD + head) * HDIM + d] = f2bf(o);
  else
    k_bf[((size_t)(head - NHEAD) * S_LEN + s) * HDIM + d] = f2bf(o);
}

// ---------------------------------------------------------------------------
// Causal flash attention. grid = (S/64 q-tiles, NH heads); block = 256.
// 8 waves: wr = 16-row M subtile, wc = 64-wide half of HD.
__global__ __launch_bounds__(256) void attn_kernel(
    const unsigned short* __restrict__ q_bf,  // [S][NH][HD]
    const unsigned short* __restrict__ k_bf,  // [NKV][S][HD]
    const unsigned short* __restrict__ vT,    // [NKV][HD][S]
    unsigned short* __restrict__ ctx) {       // [S][NH*HD]
  const int qt   = blockIdx.x;
  const int h    = blockIdx.y;
  const int kvh  = h >> 3;                    // G = NH/NKV = 8
  const int lane = threadIdx.x & 31;
  const int w    = threadIdx.x >> 5;
  const int wr   = w & 3;
  const int wc   = w >> 2;
  const int lhalf = lane >> 4;
  const int l16   = lane & 15;
  const int q0 = qt * 64 + wr * 16;
  const float scale = 0.08838834764831845f;   // HD^-0.5

  __shared__ __align__(16) unsigned short pstage[8][16 * 32];  // per-wave P tile

  // Q fragments (16 rows x 128 K, 4 fragments of K=32)
  v16bf qf[4];
  {
    const unsigned short* qrow =
        q_bf + ((size_t)(q0 + l16) * NHEAD + h) * HDIM + 8 * lhalf;
#pragma unroll
    for (int kk = 0; kk < 4; ++kk) qf[kk] = load_frag(qrow + kk * 32);
  }

  float rowM[8], rowS[8];
  v8f acc[4] = {};
#pragma unroll
  for (int r = 0; r < 8; ++r) { rowM[r] = -3.0e38f; rowS[r] = 0.0f; }

  const int nkv = (qt + 1) * 2;  // 32-wide kv tiles covering [0, qt*64+64)
  for (int kvt = 0; kvt < nkv; ++kvt) {
    const int kb = kvt * 32;

    const unsigned short* krow0 =
        k_bf + ((size_t)kvh * S_LEN + kb + l16) * HDIM + 8 * lhalf;
    const unsigned short* krow1 = krow0 + 16 * HDIM;
    const unsigned short* vbase =
        vT + ((size_t)kvh * HDIM + wc * 64 + l16) * S_LEN + kb + 8 * lhalf;

    // prefetch next KV tile (in cache by the time softmax below finishes)
    __builtin_prefetch(krow0 + 32 * HDIM, 0, 3);
    __builtin_prefetch(krow1 + 32 * HDIM, 0, 3);
    __builtin_prefetch(vbase + 32, 0, 3);

    // ---- scores: S = Q (16x128) @ K^T, two 16-col tiles.
    // Batch all 8 K fragments into distinct regs, then issue 8 WMMAs so the
    // loads overlap with matrix ops instead of serializing.
    v16bf kf0[4], kf1[4];
#pragma unroll
    for (int kk = 0; kk < 4; ++kk) {
      kf0[kk] = load_frag(krow0 + kk * 32);
      kf1[kk] = load_frag(krow1 + kk * 32);
    }
    v8f s0 = {}, s1 = {};
#pragma unroll
    for (int kk = 0; kk < 4; ++kk) {
      s0 = wmma_bf16(qf[kk], kf0[kk], s0);
      s1 = wmma_bf16(qf[kk], kf1[kk], s1);
    }

    // load this tile's V fragments early (used after softmax below)
    v16bf vf[4];
#pragma unroll
    for (int t = 0; t < 4; ++t) vf[t] = load_frag(vbase + (size_t)t * 16 * S_LEN);

    // ---- online softmax (row stats striped per C layout), stage P to LDS
    unsigned short* pst = pstage[w];
#pragma unroll
    for (int r = 0; r < 8; ++r) {
      int qpos = q0 + r + 8 * lhalf;
      int sp0 = kb + l16;
      float x0 = (sp0 <= qpos)      ? s0[r] * scale : -3.0e38f;
      float x1 = (sp0 + 16 <= qpos) ? s1[r] * scale : -3.0e38f;
      float mx = fmaxf(x0, x1);
#pragma unroll
      for (int off = 8; off > 0; off >>= 1)
        mx = fmaxf(mx, __shfl_xor(mx, off, 16));
      float newM = fmaxf(rowM[r], mx);
      float corr = __expf(rowM[r] - newM);
      float p0 = __expf(x0 - newM);
      float p1 = __expf(x1 - newM);
      float sum = p0 + p1;
#pragma unroll
      for (int off = 8; off > 0; off >>= 1)
        sum += __shfl_xor(sum, off, 16);
      rowS[r] = rowS[r] * corr + sum;
      rowM[r] = newM;
#pragma unroll
      for (int t = 0; t < 4; ++t) acc[t][r] *= corr;
      int mloc = r + 8 * lhalf;
      pst[mloc * 32 + l16]      = f2bf(p0);
      pst[mloc * 32 + 16 + l16] = f2bf(p1);
    }

    // ---- P (16x32) as A fragment: wave-internal ds store->load dependency
    v16bf pa = load_frag(pst + l16 * 32 + 8 * lhalf);

    // ---- ctx += P @ V : 4 WMMAs over this wave's 64-wide HD half
#pragma unroll
    for (int t = 0; t < 4; ++t) acc[t] = wmma_bf16(pa, vf[t], acc[t]);
  }

  // ---- epilogue: normalize and store bf16 ctx
#pragma unroll
  for (int t = 0; t < 4; ++t) {
#pragma unroll
    for (int r = 0; r < 8; ++r) {
      int m = q0 + r + 8 * lhalf;
      int n = h * HDIM + wc * 64 + t * 16 + l16;
      ctx[(size_t)m * CTXW + n] = f2bf(acc[t][r] / rowS[r]);
    }
  }
}

// ---------------------------------------------------------------------------
extern "C" void kernel_launch(void* const* d_in, const int* in_sizes, int n_in,
                              void* d_out, int out_size, void* d_ws, size_t ws_size,
                              hipStream_t stream) {
  (void)in_sizes; (void)n_in; (void)out_size; (void)ws_size;
  const float* hidden = (const float*)d_in[0];
  const int*   pos    = (const int*)d_in[1];
  const float* Wqkv   = (const float*)d_in[2];
  const float* Wo     = (const float*)d_in[3];
  const float* qnw    = (const float*)d_in[4];
  const float* knw    = (const float*)d_in[5];
  float* out = (float*)d_out;

  char* ws = (char*)d_ws;
  unsigned short* Xbf   = (unsigned short*)ws; ws += (size_t)S_LEN * 2048 * 2;
  unsigned short* WqkvT = (unsigned short*)ws; ws += (size_t)QKVW * 2048 * 2;
  unsigned short* WoT   = (unsigned short*)ws; ws += (size_t)2048 * CTXW * 2;
  float*          qkv   = (float*)ws;          ws += (size_t)S_LEN * QKVW * 4;
  unsigned short* q_bf  = (unsigned short*)ws; ws += (size_t)S_LEN * NHEAD * HDIM * 2;
  unsigned short* k_bf  = (unsigned short*)ws; ws += (size_t)NKVH * S_LEN * HDIM * 2;
  unsigned short* vTb   = (unsigned short*)ws; ws += (size_t)NKVH * HDIM * S_LEN * 2;
  unsigned short* ctxb  = (unsigned short*)ws;

  // 1) precision conversion + weight transposes
  {
    int n = S_LEN * 2048;
    cvt_kernel<<<(n + 255) / 256, 256, 0, stream>>>(hidden, Xbf, n);
  }
  {
    int n = 2048 * QKVW;
    cvtT_kernel<<<(n + 255) / 256, 256, 0, stream>>>(Wqkv, WqkvT, 2048, QKVW);
  }
  {
    int n = CTXW * 2048;
    cvtT_kernel<<<(n + 255) / 256, 256, 0, stream>>>(Wo, WoT, CTXW, 2048);
  }

  // 2) QKV projection: [2048 x 2048] @ [2048 x 5120]
  gemm_bf16<<<dim3(QKVW / 128, S_LEN / 64), 256, 0, stream>>>(
      Xbf, WqkvT, qkv, S_LEN, QKVW, 2048);

  // 3) RMSNorm + RoPE (+ V transpose)
  norm_rope_kernel<<<dim3(S_LEN, NHEAD + 2 * NKVH), 128, 0, stream>>>(
      qkv, pos, qnw, knw, q_bf, k_bf, vTb);

  // 4) causal flash attention
  attn_kernel<<<dim3(S_LEN / 64, NHEAD), 256, 0, stream>>>(q_bf, k_bf, vTb, ctxb);

  // 5) output projection: [2048 x 4096] @ [4096 x 2048]
  gemm_bf16<<<dim3(2048 / 128, S_LEN / 64), 256, 0, stream>>>(
      ctxb, WoT, out, S_LEN, 2048, CTXW);
}
